// GraphAttentionLayer_38895223832723
// MI455X (gfx1250) — matrix operand order repaired and checked by reading, not compiled, not run on hardware
//
#include <hip/hip_runtime.h>
#include <hip/hip_bf16.h>

#define NN 4096
#define DD 128
#define EPSF 1e-12f

typedef __attribute__((ext_vector_type(16))) _Float16 v16h;
typedef __attribute__((ext_vector_type(8)))  float    v8f;

// ---------------------------------------------------------------------------
// Zero fill (att matrix) — grid-stride float4 stores
// ---------------------------------------------------------------------------
__global__ __launch_bounds__(256) void zero_f4(float4* __restrict__ p, long n4) {
    long i      = (long)blockIdx.x * blockDim.x + threadIdx.x;
    long stride = (long)gridDim.x * blockDim.x;
    float4 z = make_float4(0.f, 0.f, 0.f, 0.f);
    for (; i < n4; i += stride) p[i] = z;
}

// ---------------------------------------------------------------------------
// Edge phase: one wave32 per edge. float4 per lane (D=128 = 32 lanes * 4).
// dist = ||h + case - t||; atomicAdd exp(-dist) into att[t][c].
// att (64MB) is L2-resident, so the scatter atomics stay on-chip.
// ---------------------------------------------------------------------------
__global__ __launch_bounds__(256) void edge_scatter(
    const int* __restrict__ cur, const int* __restrict__ tgt,
    const float* __restrict__ act, const float* __restrict__ cases,
    float* __restrict__ att, int E) {
    int edge = (int)((blockIdx.x * blockDim.x + threadIdx.x) >> 5);
    int lane = threadIdx.x & 31;
    if (edge >= E) return;
    int c = cur[edge];
    int t = tgt[edge];
    const float4* h4 = (const float4*)(act   + (size_t)c * DD);
    const float4* t4 = (const float4*)(act   + (size_t)t * DD);
    const float4* e4 = (const float4*)(cases + (size_t)edge * DD);
    float4 hv = h4[lane];
    float4 tv = t4[lane];
    float4 ev = e4[lane];
    float dx = hv.x + ev.x - tv.x;
    float dy = hv.y + ev.y - tv.y;
    float dz = hv.z + ev.z - tv.z;
    float dw = hv.w + ev.w - tv.w;
    float s = dx * dx + dy * dy + dz * dz + dw * dw;
    #pragma unroll
    for (int off = 16; off; off >>= 1) s += __shfl_xor(s, off, 32);
    if (lane == 0) {
        atomicAdd(att + (size_t)t * NN + c, expf(-sqrtf(s)));
    }
}

// ---------------------------------------------------------------------------
// Row sums -> inverse norms. One wave32 per row, float4 strided loads.
// ---------------------------------------------------------------------------
__global__ __launch_bounds__(256) void row_invnorm(
    const float* __restrict__ att, float* __restrict__ inv) {
    int row  = (int)((blockIdx.x * blockDim.x + threadIdx.x) >> 5);
    int lane = threadIdx.x & 31;
    if (row >= NN) return;
    const float4* r4 = (const float4*)(att + (size_t)row * NN);
    float s = 0.f;
    #pragma unroll 4
    for (int i = lane; i < NN / 4; i += 32) {
        float4 v = r4[i];
        s += v.x + v.y + v.z + v.w;
    }
    #pragma unroll
    for (int off = 16; off; off >>= 1) s += __shfl_xor(s, off, 32);
    if (lane == 0) inv[row] = 1.0f / (s + EPSF);
}

// ---------------------------------------------------------------------------
// Pre-pack a [rows, 128] fp32 row-major matrix into wave32 WMMA B-fragment
// layout in f16:
//   block (nt, kt) = 32x16 K-major tile; lane l holds N = n0 + (l&15),
//   half i holds K = k0 + (i/8)*16 + (l>>4)*8 + i%8.
//   dst offset (halves) = ((nt*ktiles + kt)*32 + lane)*16   -> 32B per lane.
// One thread per (nt, kt, lane).
// ---------------------------------------------------------------------------
__global__ __launch_bounds__(256) void pack_b16(
    const float* __restrict__ src, _Float16* __restrict__ dst, int rows) {
    int tid    = blockIdx.x * blockDim.x + threadIdx.x;
    int lane   = tid & 31;
    int blk    = tid >> 5;              // nt*ktiles + kt
    int ktiles = rows >> 5;
    if (blk >= (DD / 16) * ktiles) return;
    int nt = blk / ktiles;
    int kt = blk % ktiles;
    int lm = lane & 15;
    int hi = lane >> 4;
    const float* s = src + (size_t)(kt * 32 + hi * 8) * DD + nt * 16 + lm;
    v16h v;
    #pragma unroll
    for (int i = 0; i < 16; ++i) v[i] = (_Float16)s[(size_t)(i + (i >> 3) * 8) * DD];
    *(v16h*)(dst + ((size_t)blk * 32 + lane) * 16) = v;
}

// ---------------------------------------------------------------------------
// WMMA fragment layouts (v_wmma_f32_16x16x32_f16, wave32, per ISA 7.12.2):
//   A 16x32 f16 : lane l -> M = l&15 ; half i -> K = (i/8)*16 + (l>>4)*8 + i%8
//   B 32x16 f16 : lane l -> N = l&15 ; half i -> K = (i/8)*16 + (l>>4)*8 + i%8 (prepacked)
//   C/D 16x16 f32: vgpr j, lane l -> M = j + (l>>4)*8 , N = l&15
// ---------------------------------------------------------------------------

// GEMM1: feat[4096,128] = W[4096,4096] @ act. A streamed fp32->f16, B prepacked.
__global__ __launch_bounds__(256) void gemm_feat(
    const float* __restrict__ W, const v16h* __restrict__ actp,
    float* __restrict__ feat) {
    int wave = (int)((blockIdx.x * blockDim.x + threadIdx.x) >> 5);
    int lane = threadIdx.x & 31;
    int m0 = (wave >> 3) << 4;      // 256 M-tiles
    int nt = wave & 7;              // 8 N-tiles
    int n0 = nt << 4;
    int lm = lane & 15;
    int hi = lane >> 4;

    v8f acc = {};
    const float* arow = W + (size_t)(m0 + lm) * NN + hi * 8;
    const v16h*  bp   = actp + (size_t)nt * (NN / 32) * 32 + lane;

    for (int k0 = 0; k0 < NN; k0 += 32, bp += 32) {
        const float* ap = arow + k0;
        v16h a;
        #pragma unroll
        for (int i = 0; i < 8; ++i) {
            a[i]     = (_Float16)ap[i];        // K run  (l>>4)*8 .. +7
            a[i + 8] = (_Float16)ap[i + 16];   // K run  16 + (l>>4)*8 .. +7
        }
        v16h b = *bp;                           // contiguous 32B prepacked frag
        acc = __builtin_amdgcn_wmma_f32_16x16x32_f16(
            false, a, false, b, (short)0, acc, false, false);
    }
    #pragma unroll
    for (int j = 0; j < 8; ++j)
        feat[(size_t)(m0 + hi * 8 + j) * DD + n0 + lm] = acc[j];
}

// GEMM2: out = (att * inv_norm[row]) @ feat + feat. Row-normalization folded
// into the A-fragment load; residual folded into the epilogue; B prepacked.
__global__ __launch_bounds__(256) void gemm_out(
    const float* __restrict__ att, const float* __restrict__ inv,
    const v16h* __restrict__ featp, const float* __restrict__ feat,
    float* __restrict__ out) {
    int wave = (int)((blockIdx.x * blockDim.x + threadIdx.x) >> 5);
    int lane = threadIdx.x & 31;
    int m0 = (wave >> 3) << 4;
    int nt = wave & 7;
    int n0 = nt << 4;
    int lm = lane & 15;
    int hi = lane >> 4;

    float scale = inv[m0 + lm];
    v8f acc = {};
    const float* arow = att + (size_t)(m0 + lm) * NN + hi * 8;
    const v16h*  bp   = featp + (size_t)nt * (NN / 32) * 32 + lane;

    for (int k0 = 0; k0 < NN; k0 += 32, bp += 32) {
        const float* ap = arow + k0;
        v16h a;
        #pragma unroll
        for (int i = 0; i < 8; ++i) {
            a[i]     = (_Float16)(ap[i] * scale);
            a[i + 8] = (_Float16)(ap[i + 16] * scale);
        }
        v16h b = *bp;
        acc = __builtin_amdgcn_wmma_f32_16x16x32_f16(
            false, a, false, b, (short)0, acc, false, false);
    }
    #pragma unroll
    for (int j = 0; j < 8; ++j) {
        size_t o = (size_t)(m0 + hi * 8 + j) * DD + n0 + lm;
        out[o] = acc[j] + feat[o];
    }
}

// ---------------------------------------------------------------------------
// Host entry
// ---------------------------------------------------------------------------
extern "C" void kernel_launch(void* const* d_in, const int* in_sizes, int n_in,
                              void* d_out, int out_size, void* d_ws, size_t ws_size,
                              hipStream_t stream) {
    const int*   currents = (const int*)d_in[0];
    const int*   targets  = (const int*)d_in[1];
    const float* act      = (const float*)d_in[2];
    const float* cases    = (const float*)d_in[3];
    const float* W        = (const float*)d_in[4];
    const int E = in_sizes[0];

    float*     att   = (float*)d_ws;                          // [NN,NN]   64 MB
    float*     inv   = att + (size_t)NN * NN;                 // [NN]      16 KB
    float*     feat  = inv + NN;                              // [NN,DD]    2 MB
    _Float16*  actp  = (_Float16*)(feat + (size_t)NN * DD);   // packed B   1 MB
    _Float16*  featp = actp + (size_t)NN * DD;                // packed B   1 MB
    float*     out   = (float*)d_out;

    // 1) zero the dense attention accumulator
    zero_f4<<<4096, 256, 0, stream>>>((float4*)att, (long)NN * NN / 4);

    // 2) edge phase: dist + exp + scatter-add (memory-bound: ~530 MB streamed)
    edge_scatter<<<(E + 7) / 8, 256, 0, stream>>>(currents, targets, act, cases, att, E);

    // 3) row sums -> inverse norms
    row_invnorm<<<NN / 8, 256, 0, stream>>>(att, inv);

    // 4) pre-pack act into f16 B-fragments; GEMM1: feat = W @ act
    pack_b16<<<(DD / 16) * (NN / 32) * 32 / 256, 256, 0, stream>>>(act, actp, NN);
    gemm_feat<<<(NN / 16) * (DD / 16) / 8, 256, 0, stream>>>(W, (const v16h*)actp, feat);

    // 5) pre-pack feat; GEMM2: out = attention @ feat + feat
    pack_b16<<<(DD / 16) * (NN / 32) * 32 / 256, 256, 0, stream>>>(feat, featp, NN);
    gemm_out<<<(NN / 16) * (DD / 16) / 8, 256, 0, stream>>>(att, inv, (const v16h*)featp, feat, out);
}
